// CNNCMULT3_18674517803244
// MI455X (gfx1250) — matrix operand order repaired and checked by reading, not compiled, not run on hardware
//
#include <hip/hip_runtime.h>
#include <hip/hip_bf16.h>

typedef _Float16 h8   __attribute__((ext_vector_type(8)));
typedef _Float16 v16h __attribute__((ext_vector_type(16)));
typedef float    v8f  __attribute__((ext_vector_type(8)));

#define EPSBN 1e-5f

// ---------------- f32 -> f16 convert ----------------
__global__ void k_cvt16(const float* __restrict__ src, _Float16* __restrict__ dst, int n) {
    int i = blockIdx.x * blockDim.x + threadIdx.x;
    if (i < n) dst[i] = (_Float16)src[i];
}

// ---------------- implicit-GEMM conv via WMMA ----------------
// GEMM: M = Cout, N = 256*H*W, K = Cin*KS*KS. 'SAME' padding, stride 1.
// 256 threads (8 waves). Macro tile 64(M) x NT(N), K tile 64
// (2 WMMA K-substeps of 32 -> 2*(NT/32) v_wmma per wave per barrier interval).
// B gather: each thread owns one pixel (or half-K of one pixel) and walks K
// incrementally -> pixel decode hoisted, tap decode is add/carry only, and
// lanes read consecutive pixels -> coalesced global loads.
template<int WD, int KS, int NT>
__global__ __launch_bounds__(256)
void k_conv_wmma(const _Float16* __restrict__ in, const _Float16* __restrict__ wq,
                 float* __restrict__ out, int Cin, int Cout) {
    constexpr int H  = WD;
    constexpr int W  = WD;
    constexpr int HW = WD * WD;            // power of two: 1024 / 256 / 64
    constexpr int BS = 72;                 // padded B row stride (halfs); 144B = 9*16B
    constexpr int F  = NT / 32;            // fragments per wave along N
    constexpr int KPT = (NT * 64) / 256;   // B elems per thread (64 or 32)
    constexpr int TPN = 64 / KPT;          // threads per pixel row (1 or 2)

    __shared__ __align__(16) _Float16 sA[64 * 64];    // [m][k]   8 KB
    __shared__ __align__(16) _Float16 sB[NT * BS];    // [n][k]  padded

    const int KT   = Cin * KS * KS;
    const int Ntot = 256 * HW;
    const int nBase = blockIdx.x * NT;
    const int mBase = blockIdx.y * 64;
    const int tid   = threadIdx.x;
    const int lane  = tid & 31;
    const int wv    = tid >> 5;
    const int mi    = wv >> 1;
    const int ni    = wv & 1;
    const int laneHi = lane >> 4;
    const int lane15 = lane & 15;

    // ---- per-thread B-gather pixel decode (loop-invariant) ----
    const int nn  = (TPN == 1) ? tid : (tid >> 1);
    const int kk0 = (TPN == 1) ? 0   : ((tid & 1) * KPT);
    const int gn  = nBase + nn;
    const bool nv = gn < Ntot;
    const int bb  = gn / HW;               // shifts (HW pow2)
    const int p   = gn & (HW - 1);
    const int py  = p / W;
    const int px  = p & (W - 1);
    const _Float16* inb = in + (size_t)bb * Cin * HW;

    v8f acc[F] = {};

    for (int k0 = 0; k0 < KT; k0 += 64) {
        // ---- A tile: 64(m) x 64(k) weights, flat [Cout][Cin*KS*KS] ----
        #pragma unroll
        for (int j = 0; j < 16; ++j) {
            int idx = tid * 16 + j;              // 0..4095
            int m  = idx >> 6, kk = idx & 63;
            int gm = mBase + m, gk = k0 + kk;
            _Float16 v = (_Float16)0.f;
            if (gm < Cout && gk < KT) v = wq[gm * KT + gk];
            sA[idx] = v;
        }
        // prefetch next weight K-tile while we gather im2col (global_prefetch_b8)
        if (k0 + 64 < KT)
            __builtin_prefetch(&wq[(mBase + (tid & 63)) * KT + k0 + 64], 0, 0);

        // ---- B tile: NT(n) x 64(k) im2col, incremental tap decode ----
        {
            int gks = k0 + kk0;
            int ci, rr, ss;
            if (KS == 3) { ci = gks / 9; int r9 = gks - ci * 9; rr = r9 / 3; ss = r9 - rr * 3; }
            else         { ci = gks; rr = 0; ss = 0; }
            #pragma unroll
            for (int j0 = 0; j0 < KPT; j0 += 8) {
                h8 buf;
                #pragma unroll
                for (int jj = 0; jj < 8; ++jj) {
                    int gk = gks + j0 + jj;
                    _Float16 v = (_Float16)0.f;
                    if (KS == 3) {
                        int y = py + rr - 1, x = px + ss - 1;
                        if (nv && gk < KT && (unsigned)y < (unsigned)H && (unsigned)x < (unsigned)W)
                            v = inb[ci * HW + y * W + x];
                        if (++ss == 3) { ss = 0; if (++rr == 3) { rr = 0; ++ci; } }
                    } else {
                        if (nv && gk < KT) v = inb[ci * HW + p];
                        ++ci;
                    }
                    buf[jj] = v;
                }
                *(h8*)&sB[nn * BS + kk0 + j0] = buf;   // 16B-aligned b128 store
            }
        }
        __syncthreads();

        // ---- 2 K-substeps of 32 : per-lane fragments (ISA 7.12.2) + WMMA ----
        const int mRow = mi * 16 + lane15;
        #pragma unroll
        for (int kb = 0; kb < 64; kb += 32) {
            h8 a0 = *(const h8*)&sA[mRow * 64 + kb + laneHi * 8];
            h8 a1 = *(const h8*)&sA[mRow * 64 + kb + 16 + laneHi * 8];
            v16h a;
            #pragma unroll
            for (int t = 0; t < 8; ++t) { a[t] = a0[t]; a[t + 8] = a1[t]; }

            #pragma unroll
            for (int f = 0; f < F; ++f) {
                int col = ni * (NT / 2) + f * 16 + lane15;
                h8 b0 = *(const h8*)&sB[col * BS + kb + laneHi * 8];
                h8 b1 = *(const h8*)&sB[col * BS + kb + 16 + laneHi * 8];
                v16h bm;
                #pragma unroll
                for (int t = 0; t < 8; ++t) { bm[t] = b0[t]; bm[t + 8] = b1[t]; }
                acc[f] = __builtin_amdgcn_wmma_f32_16x16x32_f16(
                    false, a, false, bm, (short)0, acc[f], false, false);
            }
        }
        __syncthreads();
    }

    // ---- store C fragments (NCHW f32) ----
    #pragma unroll
    for (int f = 0; f < F; ++f) {
        int cgn = nBase + ni * (NT / 2) + f * 16 + lane15;
        if (cgn >= Ntot) continue;
        int cb = cgn / HW;
        int cp = cgn & (HW - 1);
        #pragma unroll
        for (int r = 0; r < 8; ++r) {
            int gm = mBase + mi * 16 + laneHi * 8 + r;
            if (gm < Cout)
                out[(cb * Cout + gm) * HW + cp] = acc[f][r];
        }
    }
}

// ---------------- per-channel batch statistics ----------------
__global__ __launch_bounds__(256)
void k_bn_stats(const float* __restrict__ x, int C, int lgHW,
                float* __restrict__ mean, float* __restrict__ invstd) {
    __shared__ float s1[256], s2[256];
    const int c = blockIdx.x;
    const int tid = threadIdx.x;
    const int HW = 1 << lgHW;
    const int total = 256 << lgHW;   // batch fixed at 256
    float s = 0.f, q = 0.f;
    for (int i = tid; i < total; i += 256) {
        int b = i >> lgHW, p = i & (HW - 1);
        float v = x[((b * C + c) << lgHW) + p];
        s += v; q += v * v;
    }
    s1[tid] = s; s2[tid] = q;
    __syncthreads();
    for (int off = 128; off > 0; off >>= 1) {
        if (tid < off) { s1[tid] += s1[tid + off]; s2[tid] += s2[tid + off]; }
        __syncthreads();
    }
    if (tid == 0) {
        float m = s1[0] / (float)total;
        float var = s2[0] / (float)total - m * m;
        mean[c] = m;
        invstd[c] = rsqrtf(var + EPSBN);
    }
}

// ---------------- BN apply + ReLU + f16 downconvert ----------------
__global__ void k_bn_relu_f16(const float* __restrict__ x, const float* __restrict__ g,
                              const float* __restrict__ bta, const float* __restrict__ mean,
                              const float* __restrict__ invstd,
                              _Float16* __restrict__ out, int C, int lgHW) {
    const int c = blockIdx.y;
    const int HW = 1 << lgHW;
    int i = blockIdx.x * 256 + threadIdx.x;      // 0 .. 256*HW-1 over (b,p)
    int b = i >> lgHW, p = i & (HW - 1);
    size_t a = ((size_t)(b * C + c) << lgHW) + p;
    float v = (x[a] - mean[c]) * invstd[c] * g[c] + bta[c];
    v = v > 0.f ? v : 0.f;
    out[a] = (_Float16)v;
}

// ---------------- 2x2 maxpool (f16) ----------------
__global__ void k_maxpool2(const _Float16* __restrict__ in, _Float16* __restrict__ out,
                           int C, int lgWo, int lgHo) {
    const int c = blockIdx.y;
    const int Wo = 1 << lgWo, Ho = 1 << lgHo;
    const int W = Wo << 1;
    int i = blockIdx.x * 256 + threadIdx.x;      // over (b, y, x)
    int x = i & (Wo - 1);
    int y = (i >> lgWo) & (Ho - 1);
    int b = i >> (lgWo + lgHo);
    const _Float16* p = &in[((size_t)(b * C + c) * (W * W)) + (y * 2) * W + x * 2];
    float v0 = (float)p[0], v1 = (float)p[1], v2 = (float)p[W], v3 = (float)p[W + 1];
    out[((size_t)(b * C + c) << (lgWo + lgHo)) + (y << lgWo) + x] =
        (_Float16)fmaxf(fmaxf(v0, v1), fmaxf(v2, v3));
}

// ---------------- final BN + ReLU + global 8x8 average ----------------
__global__ void k_bn_relu_avg(const float* __restrict__ x, const float* __restrict__ g,
                              const float* __restrict__ bta, const float* __restrict__ mean,
                              const float* __restrict__ invstd,
                              float* __restrict__ out, int C) {
    const int c = blockIdx.y;
    const int b = threadIdx.x;                   // 0..255
    const float* p = &x[(size_t)(b * C + c) << 6];
    float m = mean[c], is = invstd[c], gg = g[c], bb = bta[c];
    float s = 0.f;
    #pragma unroll 8
    for (int q = 0; q < 64; ++q) {
        float v = (p[q] - m) * is * gg + bb;
        s += v > 0.f ? v : 0.f;
    }
    out[b * C + c] = s * (1.f / 64.f);
}

extern "C" void kernel_launch(void* const* d_in, const int* in_sizes, int n_in,
                              void* d_out, int out_size, void* d_ws, size_t ws_size,
                              hipStream_t stream) {
    const float* x = (const float*)d_in[0];
    // conv biases skipped: under training-mode BatchNorm a per-channel constant
    // shifts the batch mean by exactly that constant -> xhat unchanged.
    const float* Wt[9] = { (const float*)d_in[1],  (const float*)d_in[3],  (const float*)d_in[5],
                           (const float*)d_in[7],  (const float*)d_in[9],  (const float*)d_in[11],
                           (const float*)d_in[13], (const float*)d_in[15], (const float*)d_in[17] };
    const float* G[9]; const float* Bt[9];
    for (int i = 0; i < 9; ++i) { G[i] = (const float*)d_in[19 + 2*i]; Bt[i] = (const float*)d_in[20 + 2*i]; }

    char* ws = (char*)d_ws;
    _Float16* w16  = (_Float16*)ws;                          // weight pool (~2.8 MB)
    float* mean    = (float*)(ws + (4u  << 20));
    float* invstd  = (float*)(ws + (4u  << 20) + 4096);
    _Float16* actA = (_Float16*)(ws + (8u  << 20));          // 50.4 MB max
    _Float16* actB = (_Float16*)(ws + (72u << 20));          // 50.4 MB max
    float* conv32  = (float*)(ws + (136u << 20));            // 100.7 MB max

    const int CinL[9]  = {3, 96, 96, 96, 192, 192, 192, 192, 192};
    const int CoutL[9] = {96, 96, 96, 192, 192, 192, 192, 192, 100};
    const int KSL[9]   = {3, 3, 3, 3, 3, 3, 3, 1, 1};

    size_t woff[9]; size_t o = 0;
    for (int i = 0; i < 9; ++i) { woff[i] = o; o += (size_t)CoutL[i] * CinL[i] * KSL[i] * KSL[i]; }

    for (int i = 0; i < 9; ++i) {
        int n = CoutL[i] * CinL[i] * KSL[i] * KSL[i];
        k_cvt16<<<(n + 255) / 256, 256, 0, stream>>>(Wt[i], w16 + woff[i], n);
    }
    {
        int n = 256 * 3 * 32 * 32;
        k_cvt16<<<(n + 255) / 256, 256, 0, stream>>>(x, actA, n);
    }

    _Float16* cur = actA; _Float16* nxt = actB;

    auto conv_bn = [&](auto kern, int li, int lgHW, int NT, bool last) {
        int co = CoutL[li];
        int HW = 1 << lgHW;
        int Ntot = 256 * HW;
        dim3 grid((Ntot + NT - 1) / NT, (co + 63) / 64);
        kern<<<grid, 256, 0, stream>>>(cur, w16 + woff[li], conv32, CinL[li], co);
        k_bn_stats<<<co, 256, 0, stream>>>(conv32, co, lgHW, mean, invstd);
        if (!last) {
            dim3 g2(HW, co);      // (256*HW)/256 = HW blocks per channel
            k_bn_relu_f16<<<g2, 256, 0, stream>>>(conv32, G[li], Bt[li], mean, invstd, nxt, co, lgHW);
            _Float16* t = cur; cur = nxt; nxt = t;
        } else {
            dim3 g2(1, co);
            k_bn_relu_avg<<<g2, 256, 0, stream>>>(conv32, G[li], Bt[li], mean, invstd, (float*)d_out, co);
        }
    };

    conv_bn(k_conv_wmma<32, 3, 256>, 0, 10, 256, false);
    conv_bn(k_conv_wmma<32, 3, 256>, 1, 10, 256, false);
    conv_bn(k_conv_wmma<32, 3, 256>, 2, 10, 256, false);
    {   // 32x32 -> 16x16, C=96
        dim3 g((256 * 16 * 16) / 256, 96);
        k_maxpool2<<<g, 256, 0, stream>>>(cur, nxt, 96, 4, 4);
        _Float16* t = cur; cur = nxt; nxt = t;
    }
    conv_bn(k_conv_wmma<16, 3, 256>, 3, 8, 256, false);
    conv_bn(k_conv_wmma<16, 3, 256>, 4, 8, 256, false);
    conv_bn(k_conv_wmma<16, 3, 256>, 5, 8, 256, false);
    {   // 16x16 -> 8x8, C=192
        dim3 g((256 * 8 * 8) / 256, 192);
        k_maxpool2<<<g, 256, 0, stream>>>(cur, nxt, 192, 3, 3);
        _Float16* t = cur; cur = nxt; nxt = t;
    }
    conv_bn(k_conv_wmma<8, 3, 128>, 6, 6, 128, false);
    conv_bn(k_conv_wmma<8, 1, 128>, 7, 6, 128, false);
    conv_bn(k_conv_wmma<8, 1, 128>, 8, 6, 128, true);
}
